// QuantLinear_75333726371968
// MI455X (gfx1250) — compile-verified, matrix operand order checked by reading, not run on hardware
//
#include <hip/hip_runtime.h>

typedef __attribute__((ext_vector_type(16))) _Float16 v16h;
typedef __attribute__((ext_vector_type(8)))  _Float16 v8h;
typedef __attribute__((ext_vector_type(4)))  _Float16 v4h;
typedef __attribute__((ext_vector_type(8)))  float    v8f;
typedef __attribute__((ext_vector_type(4)))  float    v4f;

#define IN_F   4096
#define OUT_F  4096
#define GROUP  128
#define BM     256            // M tile (amortizes dequant VALU over more WMMAs)
#define BN     128            // N tile
#define BK     64             // K tile (half a quant group)
#define LDSK   (BK + 8)       // 72 halves -> 144B row stride: 16B aligned, bank-skewed
#define NTILES (IN_F / BK)    // 64

// Combine two 16B LDS chunks into one v16h WMMA operand (ISA 16-bit lane layout:
// lanes 0-15 hold K {0..7,16..23}, lanes 16-31 hold K {8..15,24..31}).
__device__ __forceinline__ v16h frag16(const _Float16* row, int kb) {
    v8h lo = *(const v8h*)(row + kb);
    v8h hi = *(const v8h*)(row + kb + 16);
    return __builtin_shufflevector(lo, hi, 0, 1, 2, 3, 4, 5, 6, 7,
                                           8, 9, 10, 11, 12, 13, 14, 15);
}

// Stage one K-tile: x (f32->f16, m-major) and dequantized W (n-major) into LDS.
__device__ __forceinline__ void stage_tiles(const float* __restrict__ x,
                                            const int*   __restrict__ qw,
                                            const float* __restrict__ scales,
                                            const float* __restrict__ scales2,
                                            _Float16 (*__restrict__ A)[LDSK],
                                            _Float16 (*__restrict__ B)[LDSK],
                                            int m0, int n0, int k0, int t)
{
    // ---- x tile: 256x64 f32, 256 threads x 16 float4 ----
    const int a_kq = t & 15;              // 16 quads * 4 = 64 k
    const int a_r0 = t >> 4;              // 0..15, step 16 over 256 rows
#pragma unroll
    for (int rr = 0; rr < BM / 16; ++rr) {
        const int m = a_r0 + rr * 16;
        const v4f xv = *(const v4f*)(x + (size_t)(m0 + m) * IN_F + k0 + a_kq * 4);
        v4h hv;
        hv[0] = (_Float16)xv[0];
        hv[1] = (_Float16)xv[1];
        hv[2] = (_Float16)xv[2];
        hv[3] = (_Float16)xv[3];
        *(v4h*)&A[m][a_kq * 4] = hv;                        // ds_store_b64
    }

    // ---- weight tile: 8 qrows x 128 cols int32, dequant to f16 (N x K) ----
    const int   dq_n  = t & 127;
    const int   dq_qr = (t >> 7) * 4;     // 0 or 4
    const int   g     = k0 / GROUP;
    const float s1    = scales [(size_t)g * OUT_F + n0 + dq_n];
    const float s2    = scales2[(size_t)g * OUT_F + n0 + dq_n];
#pragma unroll
    for (int i = 0; i < 4; ++i) {
        const int qr = (k0 >> 3) + dq_qr + i;               // 8 k per int32 row
        const unsigned int q = (unsigned int)qw[(size_t)qr * OUT_F + n0 + dq_n];
        v8h hv;
#pragma unroll
        for (int j = 0; j < 8; ++j) {
            const int  nib = (q >> (4 * j)) & 15;
            const bool neg = nib < 8;
            const int  mi  = neg ? (7 - nib) : (nib - 8);
            // magnitudes*2 packed: {0,1,2,3,4,6,8,12} -> 0xC8643210
            const float mag = (float)((0xC8643210u >> (4 * mi)) & 15) * 0.5f;
            const float w   = neg ? -mag : mag;
            const float s   = neg ? s2 : s1;                // code>=0 -> scales
            hv[j] = (_Float16)(w * s);
        }
        *(v8h*)&B[dq_n][(dq_qr + i) * 8] = hv;              // ds_store_b128
    }
}

__global__ __launch_bounds__(256)
void qlinear_wmma_kernel(const float* __restrict__ x,
                         const int*   __restrict__ qw,
                         const float* __restrict__ scales,
                         const float* __restrict__ scales2,
                         const float* __restrict__ bias,
                         float* __restrict__ out)
{
    // Double-buffered tiles: A 72KB + B 36KB = 108KB of 320KB/WGP.
    __shared__ __align__(16) _Float16 Als[2][BM][LDSK];
    __shared__ __align__(16) _Float16 Bls[2][BN][LDSK];

    const int t     = threadIdx.x;        // 0..255
    const int wave  = t >> 5;             // 0..7
    const int lane  = t & 31;
    const int lrow  = lane & 15;
    const int khalf = lane >> 4;

    const int wm = wave >> 1;             // 0..3 : 64 M rows per wave
    const int wn = wave & 1;              // 0..1 : 64 N cols per wave

    const int m0 = blockIdx.y * BM;
    const int n0 = blockIdx.x * BN;

    v8f acc[4][4];                        // 64x64 per wave = 4x4 16x16 tiles
#pragma unroll
    for (int i = 0; i < 4; ++i)
#pragma unroll
        for (int j = 0; j < 4; ++j)
            acc[i][j] = (v8f)(0.0f);

    // Prologue: stage tile 0 into buffer 0.
    stage_tiles(x, qw, scales, scales2, Als[0], Bls[0], m0, n0, 0, t);
    __syncthreads();

    for (int kt = 0; kt < NTILES; ++kt) {
        const int cur = kt & 1;
        const int nxt = cur ^ 1;

        // Stage next tile into the other buffer (overlaps with WMMA below;
        // the compute section has no dependency on these loads).
        if (kt + 1 < NTILES)
            stage_tiles(x, qw, scales, scales2, Als[nxt], Bls[nxt],
                        m0, n0, (kt + 1) * BK, t);

        // ---------- WMMA over current K-tile ----------
#pragma unroll
        for (int ks = 0; ks < BK / 32; ++ks) {
            const int kb = ks * 32 + khalf * 8;
            v16h af[4];
#pragma unroll
            for (int mi = 0; mi < 4; ++mi)
                af[mi] = frag16(&Als[cur][wm * 64 + mi * 16 + lrow][0], kb);
            v16h bf[4];
#pragma unroll
            for (int ni = 0; ni < 4; ++ni)
                bf[ni] = frag16(&Bls[cur][wn * 64 + ni * 16 + lrow][0], kb);
#pragma unroll
            for (int mi = 0; mi < 4; ++mi)
#pragma unroll
                for (int ni = 0; ni < 4; ++ni)
                    acc[mi][ni] = __builtin_amdgcn_wmma_f32_16x16x32_f16(
                        /*neg_a=*/false, af[mi],
                        /*neg_b=*/false, bf[ni],
                        /*c_mod=*/(short)0, acc[mi][ni],
                        /*reuse_a=*/false, /*reuse_b=*/false);
        }

        __syncthreads();   // next-tile stores visible; current buffer free
    }

    // ---------- epilogue: C layout (VGPR r -> row r+8*khalf, col lrow) ----------
#pragma unroll
    for (int ni = 0; ni < 4; ++ni) {
        const int   n  = n0 + wn * 64 + ni * 16 + lrow;
        const float bv = bias[n];
#pragma unroll
        for (int mi = 0; mi < 4; ++mi) {
            const int m = m0 + wm * 64 + mi * 16 + khalf * 8;
#pragma unroll
            for (int r = 0; r < 8; ++r)
                out[(size_t)(m + r) * OUT_F + n] = acc[mi][ni][r] + bv;
        }
    }
}

extern "C" void kernel_launch(void* const* d_in, const int* in_sizes, int n_in,
                              void* d_out, int out_size, void* d_ws, size_t ws_size,
                              hipStream_t stream) {
    const float* x    = (const float*)d_in[0];
    const int*   qw   = (const int*)  d_in[1];
    const float* s1   = (const float*)d_in[2];
    const float* s2   = (const float*)d_in[3];
    const float* bias = (const float*)d_in[4];
    float*       out  = (float*)d_out;

    const int M = in_sizes[0] / IN_F;          // 4*2048 = 8192
    dim3 grid(OUT_F / BN, M / BM);             // (32, 32)
    qlinear_wmma_kernel<<<grid, 256, 0, stream>>>(x, qw, s1, s2, bias, out);
}
